// FQ_regression_69939247448406
// MI455X (gfx1250) — compile-verified, matrix operand order checked by reading, not run on hardware
//
#include <hip/hip_runtime.h>

typedef __attribute__((ext_vector_type(16))) _Float16 v16h;
typedef __attribute__((ext_vector_type(8)))  _Float16 v8h;
typedef __attribute__((ext_vector_type(8)))  float    v8f;

#define B_TOT  32768
#define F      64
#define R      64
#define O      64
#define KTOT   4160            // (F+1)*R
#define NCHUNK 130             // KTOT / 32
#define TILE_B 32
#define WP_OFF 65536           // byte offset of packed W in ws
#define CSTRIDE (4 * 32 * 16)  // halves per K-chunk in packed W

#define SHUF16(a, b) __builtin_shufflevector(a, b, \
    0, 1, 2, 3, 4, 5, 6, 7, 8, 9, 10, 11, 12, 13, 14, 15)

// ---------------------------------------------------------------------------
// Pack per-(f,r) params: (mean, -log2(e)/std^2, t=tanh(tt), 0.5*(1-t))
// so that exp(-((x-m)/s)^2) == exp2(d*d*q) with d = x-m, q = param.y.
// ---------------------------------------------------------------------------
__global__ void pack_params(const float* __restrict__ tt,
                            const float* __restrict__ mean,
                            const float* __restrict__ stdv,
                            float4* __restrict__ PM) {
    int i = blockIdx.x * 256 + threadIdx.x;
    if (i < F * R) {
        float t   = tanhf(tt[i]);
        float inv = 1.0f / stdv[i];
        PM[i] = make_float4(mean[i], -inv * inv * 1.4426950408889634f,
                            t, 0.5f * (1.0f - t));
    }
}

// ---------------------------------------------------------------------------
// Pre-swizzle W[O, KTOT] (f32, row-major) into f16 WMMA B-fragments.
// Fragment (chunk c, o-tile n): 32 lanes x 16 halves, lane L holds column
// o = n*16 + (L&15); half h maps to K = c*32 + kk where
//   kk = (h<8) ? (L>>4)*8 + h : 16 + (L>>4)*8 + (h-8)
// ---------------------------------------------------------------------------
__global__ void pack_W(const float* __restrict__ W, _Float16* __restrict__ Wp) {
    int tid = blockIdx.x * 256 + threadIdx.x;
    if (tid >= NCHUNK * 4 * 32) return;
    int lane = tid & 31;
    int frag = tid >> 5;
    int n = frag & 3;
    int c = frag >> 2;
    int g = lane >> 4;
    int o = n * 16 + (lane & 15);
    const float* wrow = W + o * KTOT + c * 32;
    _Float16* dst = Wp + tid * 16;
#pragma unroll
    for (int i = 0; i < 8; ++i) dst[i]     = (_Float16)wrow[g * 8 + i];
#pragma unroll
    for (int i = 0; i < 8; ++i) dst[8 + i] = (_Float16)wrow[16 + g * 8 + i];
}

// ---------------------------------------------------------------------------
// Fused kernel: 32-batch tile per block (128 threads = 4 waves).
//   Phase 1: fire[b][r] (f32 product t-norm), normalize, store w as f16.
//   Phase 2: implicit-feat GEMM; wave n owns output cols [n*16,n*16+16).
//   Per-f iteration: prefetch B fragments for f+2 (triple buffer, pinned at
//   iteration top by sched_barrier), build 4 simultaneously-live A fragments,
//   then 4 WMMAs. Xa scalars prefetched one f ahead.
// ---------------------------------------------------------------------------
__global__ __launch_bounds__(128) void fused_kernel(
        const float* __restrict__ X, const float4* __restrict__ PM,
        const _Float16* __restrict__ Wp, float* __restrict__ out) {

    __shared__ float                sXa[TILE_B][F + 8];    // f=64 is ones col
    __shared__ float                sFire[TILE_B][R + 4];
    __shared__ alignas(16) _Float16 sWgtH[TILE_B][R + 8];  // normalized w, f16
    __shared__ float                sInv[TILE_B];

    const int tid = threadIdx.x;
    const int b0  = blockIdx.x * TILE_B;

    // ---- load X tile (32x64 floats, coalesced) ----
    for (int i = tid; i < TILE_B * F; i += 128) {
        int b = i >> 6, f = i & 63;
        sXa[b][f] = X[(size_t)(b0 + b) * F + f];
    }
    if (tid < TILE_B) sXa[tid][F] = 1.0f;
    __syncthreads();

    // ---- fire[b][r]: product t-norm over features (f32, exact) ----
    {
        const int b  = tid >> 2;          // 0..31
        const int r0 = (tid & 3) * 16;    // 0,16,32,48
        float acc[16];
#pragma unroll
        for (int j = 0; j < 16; ++j) acc[j] = 1.0f;
        for (int f = 0; f < F; ++f) {
            float x = sXa[b][f];
            const float4* pm = PM + f * R + r0;
#pragma unroll
            for (int j = 0; j < 16; ++j) {
                float4 p = pm[j];                    // (mean, q, t, c)
                float d  = x - p.x;
                float gg = __builtin_amdgcn_exp2f(d * d * p.y);
                acc[j] *= fmaf(gg, p.z, p.w);        // g*t + 0.5*(1-t)
            }
        }
#pragma unroll
        for (int j = 0; j < 16; ++j) sFire[b][r0 + j] = acc[j];
    }
    __syncthreads();

    // ---- normalize over rules, store as f16 ----
    if (tid < TILE_B) {
        float s = 0.0f;
        for (int r = 0; r < R; ++r) s += sFire[tid][r];
        sInv[tid] = 1.0f / s;
    }
    __syncthreads();
    {
        const int b  = tid >> 2;
        const int r0 = (tid & 3) * 16;
        float inv = sInv[b];
#pragma unroll
        for (int j = 0; j < 16; ++j)
            sWgtH[b][r0 + j] = (_Float16)(sFire[b][r0 + j] * inv);
    }
    __syncthreads();

    // ---- implicit-feat GEMM ----
    const int wave = tid >> 5;          // o-tile index 0..3
    const int lane = tid & 31;
    const int col  = lane & 15;         // M row (tile-local) / N col
    const int g    = lane >> 4;         // lane-group for K mapping

    // hoist w fragments (f-invariant): 4 r-segments per M-tile
    const v8h w0a = *(const v8h*)&sWgtH[col][g * 8];
    const v8h w0b = *(const v8h*)&sWgtH[col][16 + g * 8];
    const v8h w0c = *(const v8h*)&sWgtH[col][32 + g * 8];
    const v8h w0d = *(const v8h*)&sWgtH[col][48 + g * 8];
    const v8h w1a = *(const v8h*)&sWgtH[16 + col][g * 8];
    const v8h w1b = *(const v8h*)&sWgtH[16 + col][16 + g * 8];
    const v8h w1c = *(const v8h*)&sWgtH[16 + col][32 + g * 8];
    const v8h w1d = *(const v8h*)&sWgtH[16 + col][48 + g * 8];

    const _Float16* wb = Wp + (size_t)(wave * 32 + lane) * 16;

    // triple-buffered B fragments: current (bf), +1 (p1), +2 loaded per iter
    v16h bf0 = *(const v16h*)(wb + (size_t)0 * CSTRIDE);
    v16h bf1 = *(const v16h*)(wb + (size_t)1 * CSTRIDE);
    v16h p1a = *(const v16h*)(wb + (size_t)2 * CSTRIDE);
    v16h p1b = *(const v16h*)(wb + (size_t)3 * CSTRIDE);

    float x0f = sXa[col][0];
    float x1f = sXa[16 + col][0];

    v8f c0 = {}, c1 = {};

#pragma unroll 2
    for (int f = 0; f <= F; ++f) {
        // prefetch B fragments for f+2 and Xa scalars for f+1; pin at top
        v16h nba = *(const v16h*)(wb + (size_t)(2 * f + 4) * CSTRIDE);
        v16h nbb = *(const v16h*)(wb + (size_t)(2 * f + 5) * CSTRIDE);
        float nx0 = sXa[col][f + 1];
        float nx1 = sXa[16 + col][f + 1];
        __builtin_amdgcn_sched_barrier(0);

        _Float16 x0 = (_Float16)x0f;
        _Float16 x1 = (_Float16)x1f;

        // all four A fragments live simultaneously -> distinct registers
        v16h a00 = SHUF16(w0a * x0, w0b * x0);   // tile0, r0=0
        v16h a10 = SHUF16(w1a * x1, w1b * x1);   // tile1, r0=0
        v16h a01 = SHUF16(w0c * x0, w0d * x0);   // tile0, r0=32
        v16h a11 = SHUF16(w1c * x1, w1d * x1);   // tile1, r0=32

        c0 = __builtin_amdgcn_wmma_f32_16x16x32_f16(
            false, a00, false, bf0, (short)0, c0, false, false);
        c1 = __builtin_amdgcn_wmma_f32_16x16x32_f16(
            false, a10, false, bf0, (short)0, c1, false, false);
        c0 = __builtin_amdgcn_wmma_f32_16x16x32_f16(
            false, a01, false, bf1, (short)0, c0, false, false);
        c1 = __builtin_amdgcn_wmma_f32_16x16x32_f16(
            false, a11, false, bf1, (short)0, c1, false, false);

        bf0 = p1a; bf1 = p1b;
        p1a = nba; p1b = nbb;
        x0f = nx0; x1f = nx1;
    }

    // ---- store C tiles: VGPR v -> row (v + 8*g), (lane&15) -> column ----
#pragma unroll
    for (int v = 0; v < 8; ++v) {
        int mo = v + g * 8;
        out[(size_t)(b0 + mo) * O + wave * 16 + col]      = c0[v];
        out[(size_t)(b0 + 16 + mo) * O + wave * 16 + col] = c1[v];
    }
}

// ---------------------------------------------------------------------------
extern "C" void kernel_launch(void* const* d_in, const int* in_sizes, int n_in,
                              void* d_out, int out_size, void* d_ws, size_t ws_size,
                              hipStream_t stream) {
    const float* X    = (const float*)d_in[0];
    const float* tt   = (const float*)d_in[1];
    const float* mean = (const float*)d_in[2];
    const float* stdv = (const float*)d_in[3];
    const float* W    = (const float*)d_in[4];

    float4*    PM = (float4*)d_ws;
    _Float16*  Wp = (_Float16*)((char*)d_ws + WP_OFF);
    float*     Yo = (float*)d_out;

    pack_params<<<(F * R + 255) / 256, 256, 0, stream>>>(tt, mean, stdv, PM);
    pack_W<<<(NCHUNK * 4 * 32 + 255) / 256, 256, 0, stream>>>(W, Wp);
    fused_kernel<<<B_TOT / TILE_B, 128, 0, stream>>>(X, PM, Wp, Yo);
}
